// CategorySpecificLinear_18674517803359
// MI455X (gfx1250) — matrix operand order repaired and checked by reading, not compiled, not run on hardware
//
#include <hip/hip_runtime.h>

typedef __attribute__((ext_vector_type(16))) __bf16 v16bf;
typedef __attribute__((ext_vector_type(8)))  __bf16 v8bf;
typedef __attribute__((ext_vector_type(2)))  __bf16 bf2;
typedef __attribute__((ext_vector_type(8)))  float  v8f;
typedef __attribute__((ext_vector_type(4)))  float  f4;
typedef __attribute__((ext_vector_type(2)))  float  f2;
typedef __attribute__((ext_vector_type(4)))  unsigned u4;
typedef __attribute__((ext_vector_type(2)))  unsigned u2;

#define D_IN    1024
#define H_OUT   1024
#define SEQ_T   32
#define KC      32      // K chunk = one bf16 WMMA K step
#define NSLAB   256     // output columns per block (8 waves x 2 tiles x 16)
#define RPITCH  40      // padded LDS row pitch in bf16 elems (80 B): conflict-free, 16B-aligned
#define RPITCHW (RPITCH / 2)   // pitch in dwords

// Split a pair of f32 into packed bf16 hi-pair and lo-pair (one dword each).
// The packed hi conversion is opaque (inline asm) so the residuals are forced to
// read its halves (v_fma_mix_f32_bf16 op_sel) instead of duplicating scalar cvts.
static __device__ __forceinline__ void split2(float f0, float f1, unsigned& h, unsigned& l) {
    unsigned hp;
    asm("v_cvt_pk_bf16_f32 %0, %1, %2" : "=v"(hp) : "v"(f0), "v"(f1));
    bf2 hv = __builtin_bit_cast(bf2, hp);
    float l0 = f0 - (float)hv[0];   // v_fma_mix_f32_bf16 op_sel lo half
    float l1 = f1 - (float)hv[1];   // v_fma_mix_f32_bf16 op_sel hi half
    bf2 lv;
    lv[0] = (__bf16)l0;             // pair -> one v_cvt_pk_bf16_f32
    lv[1] = (__bf16)l1;
    h = hp;
    l = __builtin_bit_cast(unsigned, lv);
}

// Load a 16-element bf16 fragment from LDS as two aligned 16B halves.
static __device__ __forceinline__ v16bf ld_frag(const __bf16* base, int off2) {
    v8bf a = *(const v8bf*)base;
    v8bf b = *(const v8bf*)(base + off2);
    return __builtin_shufflevector(a, b, 0,1,2,3,4,5,6,7,8,9,10,11,12,13,14,15);
}

__global__ __launch_bounds__(256, 2)
void cat_linear_wmma_kernel(const float* __restrict__ x,
                            const int*   __restrict__ cat_ids,
                            const float* __restrict__ W,
                            const float* __restrict__ bias,
                            float*       __restrict__ out)
{
    // bf16 hi/lo staged in fragment-friendly k-major layout, double buffered.
    __shared__ __bf16 sAhi[2][SEQ_T * RPITCH];   // 2 x 2.5 KB
    __shared__ __bf16 sAlo[2][SEQ_T * RPITCH];
    __shared__ __bf16 sBhi[2][NSLAB * RPITCH];   // 2 x 20 KB
    __shared__ __bf16 sBlo[2][NSLAB * RPITCH];

    const int b     = blockIdx.y;
    const int hbase = blockIdx.x * NSLAB;
    const int tid   = threadIdx.x;
    const int wave  = tid >> 5;
    const int lane  = tid & 31;
    const int n     = lane & 15;      // N inside 16x16 tile
    const int kgrp  = lane >> 4;      // which K-half this lane holds

    const int cat = cat_ids[b];
    const float* __restrict__ xb = x + (size_t)b   * SEQ_T * D_IN;
    const float* __restrict__ Wc = W + (size_t)cat * D_IN * H_OUT;

    v8f acc00 = {};  // N-tile 0, M rows 0..15
    v8f acc01 = {};  // N-tile 0, M rows 16..31
    v8f acc10 = {};  // N-tile 1, M rows 0..15
    v8f acc11 = {};  // N-tile 1, M rows 16..31

    // ---- staging assignment ----
    // A: thread -> (row, 4 consecutive k) via one float4 load
    const int ar = tid >> 3;           // 0..31
    const int ak = (tid & 7) * 4;      // 0,4,...,28
    // B: thread -> (2 consecutive cols, 16 consecutive k); b64 loads, coalesced
    const int bn = (tid & 127) * 2;    // 0,2,...,254
    const int bk = (tid >> 7) * 16;    // 0 or 16

    f4 aReg;
    f2 bReg[16];

    // preload chunk 0
    aReg = *(const f4*)(xb + (size_t)ar * D_IN + ak);
    #pragma unroll
    for (int i = 0; i < 16; ++i)
        bReg[i] = *(const f2*)(Wc + (size_t)(bk + i) * H_OUT + hbase + bn);

    int p = 0;
    for (int kb = 0; kb < D_IN; kb += KC) {
        // ---- convert + store current chunk into LDS buffer p ----
        {
            unsigned ha0, ha1, la0, la1;
            split2(aReg[0], aReg[1], ha0, la0);
            split2(aReg[2], aReg[3], ha1, la1);
            u2 tah = {ha0, ha1};
            u2 tal = {la0, la1};
            *(u2*)&sAhi[p][ar * RPITCH + ak] = tah;
            *(u2*)&sAlo[p][ar * RPITCH + ak] = tal;

            unsigned hx[8], lx[8], hy[8], ly[8];   // columns bn / bn+1, k pairs
            #pragma unroll
            for (int j = 0; j < 8; ++j) {
                split2(bReg[2 * j].x, bReg[2 * j + 1].x, hx[j], lx[j]);
                split2(bReg[2 * j].y, bReg[2 * j + 1].y, hy[j], ly[j]);
            }
            unsigned* bhp = (unsigned*)&sBhi[p][bn * RPITCH + bk];
            unsigned* blp = (unsigned*)&sBlo[p][bn * RPITCH + bk];
            u4 t;
            t = (u4){hx[0], hx[1], hx[2], hx[3]}; *(u4*)(bhp)               = t;
            t = (u4){hx[4], hx[5], hx[6], hx[7]}; *(u4*)(bhp + 4)           = t;
            t = (u4){hy[0], hy[1], hy[2], hy[3]}; *(u4*)(bhp + RPITCHW)     = t;
            t = (u4){hy[4], hy[5], hy[6], hy[7]}; *(u4*)(bhp + RPITCHW + 4) = t;
            t = (u4){lx[0], lx[1], lx[2], lx[3]}; *(u4*)(blp)               = t;
            t = (u4){lx[4], lx[5], lx[6], lx[7]}; *(u4*)(blp + 4)           = t;
            t = (u4){ly[0], ly[1], ly[2], ly[3]}; *(u4*)(blp + RPITCHW)     = t;
            t = (u4){ly[4], ly[5], ly[6], ly[7]}; *(u4*)(blp + RPITCHW + 4) = t;
        }

        // ---- issue global loads for next chunk (latency hidden behind WMMA) ----
        if (kb + KC < D_IN) {
            aReg = *(const f4*)(xb + (size_t)ar * D_IN + (kb + KC) + ak);
            #pragma unroll
            for (int i = 0; i < 16; ++i)
                bReg[i] = *(const f2*)(Wc + (size_t)(kb + KC + bk + i) * H_OUT + hbase + bn);
        }

        __syncthreads();   // buffer p fully staged

        // ---- B fragments for 2 N-tiles: lane holds k = kgrp*16 + 0..15 of col nb ----
        const int nb0 = wave * 32 + n;        // N-tile 0 column
        const int nb1 = nb0 + 16;             // N-tile 1 column
        v16bf bhi0 = ld_frag(&sBhi[p][nb0 * RPITCH + kgrp * 16], 8);
        v16bf blo0 = ld_frag(&sBlo[p][nb0 * RPITCH + kgrp * 16], 8);
        v16bf bhi1 = ld_frag(&sBhi[p][nb1 * RPITCH + kgrp * 16], 8);
        v16bf blo1 = ld_frag(&sBlo[p][nb1 * RPITCH + kgrp * 16], 8);

        // ---- A fragments per M tile; elems: k=kgrp*16+0..15 ----
        #pragma unroll
        for (int m = 0; m < 2; ++m) {
            const int row = m * 16 + n;
            v16bf ahi = ld_frag(&sAhi[p][row * RPITCH + kgrp * 16], 8);
            v16bf alo = ld_frag(&sAlo[p][row * RPITCH + kgrp * 16], 8);

            v8f c0 = m ? acc01 : acc00;
            c0 = __builtin_amdgcn_wmma_f32_16x16x32_bf16(false, ahi, false, bhi0, (short)0, c0, false, false);
            c0 = __builtin_amdgcn_wmma_f32_16x16x32_bf16(false, ahi, false, blo0, (short)0, c0, false, false);
            c0 = __builtin_amdgcn_wmma_f32_16x16x32_bf16(false, alo, false, bhi0, (short)0, c0, false, false);
            if (m) acc01 = c0; else acc00 = c0;

            v8f c1 = m ? acc11 : acc10;
            c1 = __builtin_amdgcn_wmma_f32_16x16x32_bf16(false, ahi, false, bhi1, (short)0, c1, false, false);
            c1 = __builtin_amdgcn_wmma_f32_16x16x32_bf16(false, ahi, false, blo1, (short)0, c1, false, false);
            c1 = __builtin_amdgcn_wmma_f32_16x16x32_bf16(false, alo, false, bhi1, (short)0, c1, false, false);
            if (m) acc11 = c1; else acc10 = c1;
        }

        p ^= 1;   // next chunk stages into the other buffer (no extra barrier needed)
    }

    // ---- epilogue: bias add + store (C/D layout: VGPR r -> M = r + 8*kgrp) ----
    const int hw0 = hbase + wave * 32 + n;
    const int hw1 = hw0 + 16;
    const float bv0 = bias[(size_t)cat * H_OUT + hw0];
    const float bv1 = bias[(size_t)cat * H_OUT + hw1];
    float* __restrict__ ob0 = out + (size_t)b * SEQ_T * H_OUT + hw0;
    float* __restrict__ ob1 = out + (size_t)b * SEQ_T * H_OUT + hw1;
    #pragma unroll
    for (int r = 0; r < 8; ++r) {
        int mrow = r + kgrp * 8;
        ob0[(size_t)mrow * H_OUT]        = acc00[r] + bv0;
        ob0[(size_t)(16 + mrow) * H_OUT] = acc01[r] + bv0;
        ob1[(size_t)mrow * H_OUT]        = acc10[r] + bv1;
        ob1[(size_t)(16 + mrow) * H_OUT] = acc11[r] + bv1;
    }
}

extern "C" void kernel_launch(void* const* d_in, const int* in_sizes, int n_in,
                              void* d_out, int out_size, void* d_ws, size_t ws_size,
                              hipStream_t stream) {
    (void)in_sizes; (void)n_in; (void)out_size; (void)d_ws; (void)ws_size;
    const float* x       = (const float*)d_in[0];
    const int*   cat_ids = (const int*)  d_in[1];
    const float* W       = (const float*)d_in[2];
    const float* bias    = (const float*)d_in[3];
    float*       out     = (float*)d_out;

    dim3 grid(H_OUT / NSLAB, /*B=*/256, 1);   // (4, 256)
    dim3 block(256, 1, 1);                     // 8 waves of 32
    cat_linear_wmma_kernel<<<grid, block, 0, stream>>>(x, cat_ids, W, bias, out);
}
